// SparseAttention_83949430767989
// MI455X (gfx1250) — compile-verified, matrix operand order checked by reading, not run on hardware
//
#include <hip/hip_runtime.h>
#include <hip/hip_bf16.h>
#include <stdint.h>

typedef __attribute__((ext_vector_type(16))) __bf16       v16bf;
typedef __attribute__((ext_vector_type(8)))  float        v8f;
typedef __attribute__((ext_vector_type(4)))  unsigned int v4u;

#define WMMA_BF16(a, b, c) \
  __builtin_amdgcn_wmma_f32_16x16x32_bf16(false, (a), false, (b), (short)0, (c), false, false)

static constexpr int BATCH = 2;
static constexpr int SEQ   = 2048;
static constexpr int HID   = 1024;
static constexpr int NH    = 16;
static constexpr int HD    = 64;
static constexpr int MROWS = BATCH * SEQ;   // 4096

// ---------------------------------------------------------------------------
// Fragment loader: 16-bit A/B fragment for v_wmma_*_16x16x32_bf16.
// ISA 7.12.2: lanes 0-15 hold K = {0..7, 16..23}, lanes 16-31 K = {8..15, 24..31}
// => per lane: two contiguous 8-element (16B) runs at +8*half and +16+8*half.
// Works for both global and LDS pointers (ds_load_b128 for shared).
// ---------------------------------------------------------------------------
__device__ __forceinline__ v16bf load_frag(const __bf16* row, int hf) {
  union U { v4u q[2]; v16bf v; } u;
  u.q[0] = *(const v4u*)(row + 8 * hf);
  u.q[1] = *(const v4u*)(row + 16 + 8 * hf);
  return u.v;
}

__device__ __forceinline__ v8f vzero8() {
  v8f z;
#pragma unroll
  for (int i = 0; i < 8; ++i) z[i] = 0.0f;
  return z;
}

// ---------------------------------------------------------------------------
// fp32 -> bf16 elementwise convert
// ---------------------------------------------------------------------------
__global__ void f2bf(const float* __restrict__ in, __bf16* __restrict__ out, int n) {
  int i = blockIdx.x * blockDim.x + threadIdx.x;
  if (i < n) out[i] = (__bf16)in[i];
}

// ---------------------------------------------------------------------------
// bool[SEQ][SEQ] -> bitmask u32[SEQ][SEQ/32]
// ---------------------------------------------------------------------------
__global__ void build_maskbits(const unsigned char* __restrict__ m,
                               uint32_t* __restrict__ bits, int nwords) {
  int w = blockIdx.x * blockDim.x + threadIdx.x;
  if (w >= nwords) return;
  const unsigned char* p = m + (size_t)w * 32;
  uint32_t v = 0;
#pragma unroll
  for (int i = 0; i < 32; ++i) v |= (uint32_t)(p[i] != 0) << i;
  bits[w] = v;
}

// ---------------------------------------------------------------------------
// WMMA GEMM:  out = A[M=4096,K=1024](bf16) @ W[N=1024,K=1024]^T (bf16) + bias
// One wave computes a 16(M) x 64(N) tile.
// mode 0: store bf16 [B,H,S,D]        (Q, K)
// mode 1: store bf16 [B,H,D,S]        (V, transposed for the PV matmul)
// mode 2: store fp32 [M,HID]          (final output projection)
// ---------------------------------------------------------------------------
__global__ void __launch_bounds__(128) gemm16(const __bf16* __restrict__ A,
                                              const __bf16* __restrict__ W,
                                              const float* __restrict__ bias,
                                              void* __restrict__ outp, int mode) {
  int wave = blockIdx.x * (blockDim.x >> 5) + (threadIdx.x >> 5);
  int lane = threadIdx.x & 31;
  int hf = lane >> 4, ln = lane & 15;
  const int NTILES = HID / 64;        // 16
  int mt = wave / NTILES;             // 0..255
  int nt = wave % NTILES;             // 0..15

  const __bf16* arow = A + (size_t)(mt * 16 + ln) * HID;
  v8f acc[4];
#pragma unroll
  for (int s = 0; s < 4; ++s) acc[s] = vzero8();

  for (int kb = 0; kb < HID; kb += 32) {
    v16bf af = load_frag(arow + kb, hf);
#pragma unroll
    for (int s = 0; s < 4; ++s) {
      int n = nt * 64 + s * 16 + ln;
      v16bf bf_ = load_frag(W + (size_t)n * HID + kb, hf);
      acc[s] = WMMA_BF16(af, bf_, acc[s]);
    }
  }

  // C/D layout: lane ln holds column n, rows (r + 8*hf), r = 0..7
  int mbase = mt * 16 + 8 * hf;
  int bglob = mbase / SEQ;
  int sbase = mbase % SEQ;

  if (mode == 0) {                       // bf16 [B,H,S,D]
    __bf16* out = (__bf16*)outp;
#pragma unroll
    for (int s = 0; s < 4; ++s) {
      int n = nt * 64 + s * 16 + ln;
      int h = n >> 6, d = n & 63;
      float bb = bias[n];
      size_t base = (((size_t)(bglob * NH + h)) * SEQ + sbase) * HD + d;
#pragma unroll
      for (int r = 0; r < 8; ++r)
        out[base + (size_t)r * HD] = (__bf16)(acc[s][r] + bb);
    }
  } else if (mode == 1) {                // bf16 [B,H,D,S]  (V transposed)
    __bf16* out = (__bf16*)outp;
#pragma unroll
    for (int s = 0; s < 4; ++s) {
      int n = nt * 64 + s * 16 + ln;
      int h = n >> 6, d = n & 63;
      float bb = bias[n];
      union { v4u q; __bf16 e[8]; } pk;
#pragma unroll
      for (int r = 0; r < 8; ++r) pk.e[r] = (__bf16)(acc[s][r] + bb);
      size_t base = (((size_t)(bglob * NH + h)) * HD + d) * SEQ + sbase;
      *(v4u*)(out + base) = pk.q;        // 8 consecutive s per lane: 16B store
    }
  } else {                               // fp32 [M,HID]
    float* out = (float*)outp;
#pragma unroll
    for (int s = 0; s < 4; ++s) {
      int n = nt * 64 + s * 16 + ln;
      float bb = bias[n];
#pragma unroll
      for (int r = 0; r < 8; ++r)
        out[(size_t)(mbase + r) * HID + n] = acc[s][r] + bb;
    }
  }
}

// ---------------------------------------------------------------------------
// Flash-style sparse attention. 1 wave = 16 query rows of one (b,h).
// Q,K: bf16 [B,H,S,D]; Vt: bf16 [B,H,D,S]; ctx out: bf16 [B,S,HID]
// K tiles (32 keys x 64d = 4KB, contiguous) are double-buffered into LDS via
// the CDNA5 async global->LDS engine (ASYNCcnt) so the DMA for tile j+32
// overlaps the WMMAs of tile j.
// ---------------------------------------------------------------------------
__global__ void __launch_bounds__(32) sparse_attn(const __bf16* __restrict__ Q,
                                                  const __bf16* __restrict__ K,
                                                  const __bf16* __restrict__ Vt,
                                                  const uint32_t* __restrict__ mbits,
                                                  __bf16* __restrict__ ctx) {
  __shared__ __bf16 kbuf[2][32][64];  // double-buffered K tile (2 x 4KB)
  __shared__ __bf16 pbuf[16][32];     // P tile: C-layout -> A-layout reshuffle
  int lane = threadIdx.x & 31;
  int hf = lane >> 4, ln = lane & 15;
  int qt = blockIdx.x & (SEQ / 16 - 1);
  int bh = blockIdx.x >> 7;
  int q0 = qt * 16;

  const __bf16* Qp = Q + (size_t)bh * SEQ * HD;
  const __bf16* Kp = K + (size_t)bh * SEQ * HD;
  const __bf16* Vp = Vt + (size_t)bh * HD * SEQ;

  const __bf16* qrow = Qp + (size_t)(q0 + ln) * HD;
  v16bf a_lo = load_frag(qrow, hf);        // d = 0..31
  v16bf a_hi = load_frag(qrow + 32, hf);   // d = 32..63

  // LDS byte offset of kbuf (flat LDS aperture: low 32 bits are the DS offset)
  uint32_t kbase = (uint32_t)(size_t)(&kbuf[0][0][0]);

  // Issue 8 x global_load_async_to_lds_b128 (32 lanes x 16B = 512B each) to
  // copy one contiguous 4KB K tile (keys j..j+31) into LDS buffer `buf`.
  auto issue_ktile = [&](int buf, int j) {
    unsigned long long g = (unsigned long long)(size_t)(Kp + (size_t)j * HD);
#pragma unroll
    for (int c = 0; c < 8; ++c) {
      unsigned int       la = kbase + (unsigned)buf * 4096u + (unsigned)(lane * 16 + c * 512);
      unsigned long long ga = g + (unsigned long long)(lane * 16 + c * 512);
      asm volatile("global_load_async_to_lds_b128 %0, %1, off"
                   :: "v"(la), "v"(ga) : "memory");
    }
  };

  float m8[8], l8[8], corr[8];
  v8f o[4];
#pragma unroll
  for (int r = 0; r < 8; ++r) { m8[r] = -1e30f; l8[r] = 0.0f; }
#pragma unroll
  for (int t = 0; t < 4; ++t) o[t] = vzero8();

  const float sc = 0.125f;                 // 1/sqrt(64)

  issue_ktile(0, 0);
  int cur = 0;

  for (int j = 0; j < SEQ; j += 32) {
    bool more = (j + 32) < SEQ;
    if (more) {
      issue_ktile(cur ^ 1, j + 32);        // prefetch next tile via async DMA
      asm volatile("s_wait_asynccnt 0x8" ::: "memory");  // current tile ready
    } else {
      asm volatile("s_wait_asynccnt 0x0" ::: "memory");
    }

    // stream next V rows through L2 while we compute
    __builtin_prefetch(Vp + (size_t)lane * SEQ + j + 32, 0, 0);
    __builtin_prefetch(Vp + (size_t)(lane + 32) * SEQ + j + 32, 0, 0);

    const __bf16* kr0 = &kbuf[cur][ln][0];        // keys j+ln
    const __bf16* kr1 = &kbuf[cur][ln + 16][0];   // keys j+16+ln

    v8f z = vzero8();
    v8f s0 = WMMA_BF16(a_lo, load_frag(kr0, hf), z);
    s0 = WMMA_BF16(a_hi, load_frag(kr0 + 32, hf), s0);
    v8f s1 = WMMA_BF16(a_lo, load_frag(kr1, hf), z);
    s1 = WMMA_BF16(a_hi, load_frag(kr1 + 32, hf), s1);

#pragma unroll
    for (int r = 0; r < 8; ++r) {
      int row = q0 + r + 8 * hf;
      uint32_t w = mbits[(size_t)row * (SEQ / 32) + (j >> 5)];
      float x0 = ((w >> ln) & 1u) ? s0[r] * sc : -1e9f;
      float x1 = ((w >> (ln + 16)) & 1u) ? s1[r] * sc : -1e9f;
      // row max across the 16 lanes of this half (wave32, xor<16 stays in half)
      float tv = fmaxf(x0, x1);
      tv = fmaxf(tv, __shfl_xor(tv, 1));
      tv = fmaxf(tv, __shfl_xor(tv, 2));
      tv = fmaxf(tv, __shfl_xor(tv, 4));
      tv = fmaxf(tv, __shfl_xor(tv, 8));
      float mn = fmaxf(m8[r], tv);
      float cf = __expf(m8[r] - mn);
      float p0 = __expf(x0 - mn), p1 = __expf(x1 - mn);
      float rs = p0 + p1;
      rs += __shfl_xor(rs, 1);
      rs += __shfl_xor(rs, 2);
      rs += __shfl_xor(rs, 4);
      rs += __shfl_xor(rs, 8);
      l8[r] = l8[r] * cf + rs;
      m8[r] = mn;
      corr[r] = cf;
      pbuf[r + 8 * hf][ln]      = (__bf16)p0;
      pbuf[r + 8 * hf][ln + 16] = (__bf16)p1;
    }
#pragma unroll
    for (int t = 0; t < 4; ++t)
#pragma unroll
      for (int r = 0; r < 8; ++r) o[t][r] *= corr[r];

    __syncthreads();
    v16bf ap = load_frag(&pbuf[ln][0], hf);   // P as A-fragment (16q x 32k)
#pragma unroll
    for (int t = 0; t < 4; ++t) {
      const __bf16* vrow = Vp + (size_t)(t * 16 + ln) * SEQ + j;  // contiguous keys
      o[t] = WMMA_BF16(ap, load_frag(vrow, hf), o[t]);
    }
    __syncthreads();
    cur ^= 1;
  }

  int b = bh >> 4, h = bh & 15;
#pragma unroll
  for (int r = 0; r < 8; ++r) {
    float inv = 1.0f / l8[r];
    size_t base = ((size_t)b * SEQ + q0 + r + 8 * hf) * HID + h * HD;
#pragma unroll
    for (int t = 0; t < 4; ++t)
      ctx[base + t * 16 + ln] = (__bf16)(o[t][r] * inv);
  }
}

// ---------------------------------------------------------------------------
extern "C" void kernel_launch(void* const* d_in, const int* in_sizes, int n_in,
                              void* d_out, int out_size, void* d_ws, size_t ws_size,
                              hipStream_t stream) {
  const float* X  = (const float*)d_in[0];
  const float* Wq = (const float*)d_in[1];
  const float* bq = (const float*)d_in[2];
  const float* Wk = (const float*)d_in[3];
  const float* bk = (const float*)d_in[4];
  const float* Wv = (const float*)d_in[5];
  const float* bv = (const float*)d_in[6];
  const float* Wo = (const float*)d_in[7];
  const float* bo = (const float*)d_in[8];
  const unsigned char* mask = (const unsigned char*)d_in[9];

  char* ws = (char*)d_ws;
  size_t off = 0;
  auto alloc = [&](size_t bytes) {
    void* p = ws + off;
    off = (off + bytes + 255) & ~(size_t)255;
    return p;
  };
  __bf16* Xb   = (__bf16*)alloc((size_t)MROWS * HID * 2);
  __bf16* Wqb  = (__bf16*)alloc((size_t)HID * HID * 2);
  __bf16* Wkb  = (__bf16*)alloc((size_t)HID * HID * 2);
  __bf16* Wvb  = (__bf16*)alloc((size_t)HID * HID * 2);
  __bf16* Wob  = (__bf16*)alloc((size_t)HID * HID * 2);
  __bf16* Qw   = (__bf16*)alloc((size_t)MROWS * HID * 2);
  __bf16* Kw   = (__bf16*)alloc((size_t)MROWS * HID * 2);
  __bf16* Vtw  = (__bf16*)alloc((size_t)MROWS * HID * 2);
  __bf16* Ctx  = (__bf16*)alloc((size_t)MROWS * HID * 2);
  uint32_t* Mb = (uint32_t*)alloc((size_t)SEQ * (SEQ / 32) * 4);

  const int nX = MROWS * HID;          // 4194304
  const int nW = HID * HID;            // 1048576
  f2bf<<<(nX + 255) / 256, 256, 0, stream>>>(X, Xb, nX);
  f2bf<<<(nW + 255) / 256, 256, 0, stream>>>(Wq, Wqb, nW);
  f2bf<<<(nW + 255) / 256, 256, 0, stream>>>(Wk, Wkb, nW);
  f2bf<<<(nW + 255) / 256, 256, 0, stream>>>(Wv, Wvb, nW);
  f2bf<<<(nW + 255) / 256, 256, 0, stream>>>(Wo, Wob, nW);

  const int nwords = SEQ * (SEQ / 32); // 131072
  build_maskbits<<<(nwords + 255) / 256, 256, 0, stream>>>(mask, Mb, nwords);

  // 4096 waves per GEMM, 4 waves / block
  const int gemmBlocks = (MROWS / 16) * (HID / 64) / 4;  // 1024
  gemm16<<<gemmBlocks, 128, 0, stream>>>(Xb, Wqb, bq, (void*)Qw, 0);
  gemm16<<<gemmBlocks, 128, 0, stream>>>(Xb, Wkb, bk, (void*)Kw, 0);
  gemm16<<<gemmBlocks, 128, 0, stream>>>(Xb, Wvb, bv, (void*)Vtw, 1);

  sparse_attn<<<BATCH * NH * (SEQ / 16), 32, 0, stream>>>(Qw, Kw, Vtw, Mb, Ctx);

  gemm16<<<gemmBlocks, 128, 0, stream>>>(Ctx, Wob, bo, d_out, 2);
}
